// Mamba3_40578851013190
// MI455X (gfx1250) — matrix operand order repaired and checked by reading, not compiled, not run on hardware
//
#include <hip/hip_runtime.h>
#include <hip/hip_bf16.h>

#define DEVINL __device__ __forceinline__

typedef __attribute__((ext_vector_type(16))) __bf16 v16bf;
typedef __attribute__((ext_vector_type(8)))  __bf16 v8bf;
typedef __attribute__((ext_vector_type(8)))  float  v8f;

constexpr int B_   = 2;
constexpr int L_   = 1024;
constexpr int HID  = 2048;
constexpr int N_   = 128;
constexpr int H_   = 64;
constexpr int P_   = 64;
constexpr int NA_  = 32;
constexpr int DIN  = 2*H_*P_ + 2*N_ + 3*H_ + NA_;   // 8672
constexpr int ROWS = B_*L_;                         // 2048

// split offsets inside one proj row
constexpr int OFF_Z    = 0;
constexpr int OFF_X    = H_*P_;          // 4096
constexpr int OFF_B    = 2*H_*P_;        // 8192
constexpr int OFF_C    = OFF_B + N_;     // 8320
constexpr int OFF_DT   = OFF_C + N_;     // 8448
constexpr int OFF_A    = OFF_DT + H_;    // 8512
constexpr int OFF_TRAP = OFF_A + H_;     // 8576
constexpr int OFF_ANG  = OFF_TRAP + H_;  // 8640

// ---------------------------------------------------------------- converts
__global__ void k_f32_to_bf16(const float* __restrict__ in, __bf16* __restrict__ out, long n) {
  long i = (long)blockIdx.x*blockDim.x + threadIdx.x;
  long stride = (long)gridDim.x*blockDim.x;
  for (; i < n; i += stride) out[i] = (__bf16)in[i];
}

// ---------------------------------------------------------------- WMMA GEMM
// C[M,N] = A[M,K] (row-major bf16) @ B[N,K]^T (row-major bf16, i.e. weight rows)
// Block tile 128x128, 8 waves, wave tile 32x64 (2x4 WMMA 16x16x32 tiles).
// Tiles staged via GLOBAL_LOAD_ASYNC_TO_LDS_B128 with LDS double buffering.
DEVINL v16bf cat8(v8bf a, v8bf b) {
  return __builtin_shufflevector(a, b, 0,1,2,3,4,5,6,7,8,9,10,11,12,13,14,15);
}

DEVINL void async_ld_b128(unsigned lds_off, const void* gaddr) {
  asm volatile("global_load_async_to_lds_b128 %0, %1, off"
               :: "v"(lds_off), "v"((unsigned long long)gaddr) : "memory");
}
DEVINL void wait_async0() {
  asm volatile("s_wait_asynccnt 0" ::: "memory");
}

constexpr int BK    = 32;
constexpr int LDS_S = BK + 8;   // pad (80B row pitch, 16B-aligned) to dodge bank conflicts

__global__ __launch_bounds__(256)
void k_gemm_bf16(const __bf16* __restrict__ A, const __bf16* __restrict__ Bm,
                 float* __restrict__ C, int M, int N, int K)
{
  __shared__ __align__(16) __bf16 As[2][128][LDS_S];
  __shared__ __align__(16) __bf16 Bs[2][128][LDS_S];
  const int tid  = threadIdx.x;
  const int lane = tid & 31;
  const int wave = tid >> 5;
  const int m0 = blockIdx.y * 128;
  const int n0 = blockIdx.x * 128;
  const int wm = (wave >> 1) * 32;   // wave rows: 0/32/64/96
  const int wn = (wave & 1) * 64;    // wave cols: 0/64

  const unsigned asbase = (unsigned)(uintptr_t)(&As[0][0][0]);
  const unsigned bsbase = (unsigned)(uintptr_t)(&Bs[0][0][0]);
  constexpr unsigned BUFB = 128u * LDS_S * 2u;  // bytes per buffer

  v8f acc[2][4];
#pragma unroll
  for (int i = 0; i < 2; i++)
#pragma unroll
    for (int j = 0; j < 4; j++) { v8f z = {}; acc[i][j] = z; }

  const int lrow = tid >> 2;        // 0..63
  const int lcol = (tid & 3) * 8;   // bf16 units: 0,8,16,24

  // async-stage one 128x32 A slab + one 128x32 B slab into buffer `buf`
  auto stage = [&](int buf, int k0) {
#pragma unroll
    for (int i = 0; i < 2; i++) {
      const int row = lrow + i * 64;
      const unsigned loff = (unsigned)buf * BUFB + (unsigned)(row * LDS_S + lcol) * 2u;
      // A rows: M is a multiple of 128 for this problem -> always in range
      async_ld_b128(asbase + loff, A + (long)(m0 + row) * K + k0 + lcol);
      // B rows: clamp ragged edge; garbage only feeds masked-out output columns
      int gn = n0 + row; gn = (gn < N) ? gn : (N - 1);
      async_ld_b128(bsbase + loff, Bm + (long)gn * K + k0 + lcol);
    }
  };

  const int nslab = K / BK;
  stage(0, 0);
  wait_async0();
  __syncthreads();

  const int ar   = lane & 15;
  const int aklo = (lane < 16) ? 0 : 8;    // A: lanes 0-15 -> K{0..7,16..23}; 16-31 -> K{8..15,24..31}
  const int bklo = (lane < 16) ? 0 : 8;    // B: lanes 0-15 -> K{0..7,16..23}; 16-31 -> K{8..15,24..31}

  for (int s = 0; s < nslab; s++) {
    const int cur = s & 1;
    if (s + 1 < nslab) stage(cur ^ 1, (s + 1) * BK);

    v16bf af[2], bfr[4];
#pragma unroll
    for (int i = 0; i < 2; i++) {
      const __bf16* ap = &As[cur][wm + i*16 + ar][aklo];
      af[i] = cat8(*(const v8bf*)ap, *(const v8bf*)(ap + 16));
    }
#pragma unroll
    for (int j = 0; j < 4; j++) {
      const __bf16* bp = &Bs[cur][wn + j*16 + ar][bklo];
      bfr[j] = cat8(*(const v8bf*)bp, *(const v8bf*)(bp + 16));
    }
#pragma unroll
    for (int i = 0; i < 2; i++)
#pragma unroll
      for (int j = 0; j < 4; j++)
        acc[i][j] = __builtin_amdgcn_wmma_f32_16x16x32_bf16(
            false, af[i], false, bfr[j], (short)0, acc[i][j], false, false);

    wait_async0();       // next slab landed in the other buffer
    __syncthreads();     // + everyone done reading `cur` before it is restaged
  }

  // D layout: VGPR r -> M = r (lanes 0-15) / r+8 (lanes 16-31), N = lane&15
  const int cr   = lane & 15;
  const int rsel = (lane >> 4) << 3;
#pragma unroll
  for (int i = 0; i < 2; i++)
#pragma unroll
    for (int j = 0; j < 4; j++) {
      int colg = n0 + wn + j*16 + cr;
      if (colg >= N) continue;
#pragma unroll
      for (int r = 0; r < 8; r++) {
        int rowg = m0 + wm + i*16 + r + rsel;
        C[(long)rowg*N + colg] = acc[i][j][r];
      }
    }
}

// ---------------------------------------------------------------- prep
DEVINL float softplusf_(float x) { return (x > 20.f) ? x : log1pf(expf(x)); }
DEVINL float sigmoidf_(float x)  { return 1.f / (1.f + expf(-x)); }

__global__ __launch_bounds__(256)
void k_prep(const float* __restrict__ proj, const float* __restrict__ dt_bias,
            const float* __restrict__ Bnw, const float* __restrict__ Cnw,
            float* __restrict__ DTb, float* __restrict__ decb, float* __restrict__ lamb,
            float* __restrict__ ub, float* __restrict__ Brms, float* __restrict__ Crms)
{
  const int row = blockIdx.x;
  const int tid = threadIdx.x;
  const float* pr = proj + (long)row * DIN;
  __shared__ float red[256];
  __shared__ float sDT[H_];

  float v = (tid < 128) ? pr[OFF_B + tid] : pr[OFF_C + (tid - 128)];
  red[tid] = v * v;
  __syncthreads();
  const int l = tid & 127, half = tid & 128;
  for (int s = 64; s > 0; s >>= 1) {
    if (l < s) red[half + l] += red[half + l + s];
    __syncthreads();
  }
  if (tid < 128) {
    float rs = rsqrtf(red[0] * (1.f/128.f) + 1e-5f);
    Brms[(long)row*N_ + tid] = v * rs * Bnw[tid];
  } else {
    float rs = rsqrtf(red[128] * (1.f/128.f) + 1e-5f);
    Crms[(long)row*N_ + (tid-128)] = v * rs * Cnw[tid-128];
  }
  if (tid < H_) {
    float dt = softplusf_(pr[OFF_DT + tid] + dt_bias[tid]);
    float Av = fminf(-softplusf_(pr[OFF_A + tid]), -1e-4f);
    DTb [(long)row*H_ + tid] = dt;
    decb[(long)row*H_ + tid] = expf(Av * dt);
    lamb[(long)row*H_ + tid] = sigmoidf_(pr[OFF_TRAP + tid]);
    sDT[tid] = dt;
  }
  __syncthreads();
#pragma unroll
  for (int j = 0; j < 16; j++) {
    int idx = tid * 16 + j;                         // 256*16 = 4096
    ub[(long)row*(H_*P_) + idx] = pr[OFF_X + idx] * sDT[idx >> 6];
  }
}

// ---------------------------------------------------------------- scan
__global__ __launch_bounds__(256)
void k_scan(const float* __restrict__ proj,
            const float* __restrict__ DTb, const float* __restrict__ decb,
            const float* __restrict__ lamb, const float* __restrict__ ub,
            const float* __restrict__ Brms, const float* __restrict__ Crms,
            const float* __restrict__ B_bias, const float* __restrict__ C_bias,
            float* __restrict__ yb)
{
  const int bh  = blockIdx.x;      // b*H + h
  const int b   = bh >> 6;
  const int h   = bh & 63;
  const int tid = threadIdx.x;
  const int p     = tid >> 2;          // 0..63
  const int nbase = (tid & 3) * 32;    // n slice

  __shared__ float sPhase[NA_], sCos[NA_], sSin[NA_];
  __shared__ float sB[N_], sC[N_], sRB[N_], sRC[N_];

  if (tid < NA_) sPhase[tid] = 0.f;

  float hreg[32], gpreg[32];
#pragma unroll
  for (int i = 0; i < 32; i++) { hreg[i] = 0.f; gpreg[i] = 0.f; }

  for (int t = 0; t < L_; t++) {
    const long row = (long)b * L_ + t;
    // stage 1: phase cumsum + raw B/C (+bias)
    if (tid < NA_) {
      float dt = DTb[row*H_ + h];
      float ph = sPhase[tid] + proj[row*DIN + OFF_ANG + tid] * dt;
      sPhase[tid] = ph;
      float sn, cs; sincosf(ph, &sn, &cs);
      sCos[tid] = cs; sSin[tid] = sn;
    }
    if (tid < N_) sRB[tid]      = Brms[row*N_ + tid]        + B_bias[h*N_ + tid];
    else          sRC[tid - N_] = Crms[row*N_ + (tid - N_)] + C_bias[h*N_ + (tid - N_)];
    __syncthreads();
    // stage 2: rope
    if (tid < N_) {
      float o;
      if      (tid < NA_)   o = sRB[tid]      * sCos[tid]      - sRB[tid + NA_] * sSin[tid];
      else if (tid < 2*NA_) o = sRB[tid - NA_]* sSin[tid - NA_]+ sRB[tid]       * sCos[tid - NA_];
      else                  o = sRB[tid];
      sB[tid] = o;
    } else {
      int n = tid - N_;
      float o;
      if      (n < NA_)   o = sRC[n]      * sCos[n]      - sRC[n + NA_] * sSin[n];
      else if (n < 2*NA_) o = sRC[n - NA_]* sSin[n - NA_]+ sRC[n]       * sCos[n - NA_];
      else                o = sRC[n];
      sC[n] = o;
    }
    __syncthreads();
    // stage 3: recurrence + contraction
    float dec = decb[row*H_ + h];
    float lm  = lamb[row*H_ + h];
    float up  = ub[row*(H_*P_) + h*P_ + p];
    float c1 = 1.f - lm;
    float c2 = lm * dec;
    float acc = 0.f;
#pragma unroll
    for (int i = 0; i < 32; i++) {
      float Bn = sB[nbase + i];
      float G  = up * Bn;
      float hv = fmaf(dec, hreg[i], fmaf(c1, G, c2 * gpreg[i]));
      hreg[i]  = hv;
      gpreg[i] = G;
      acc = fmaf(hv, sC[nbase + i], acc);
    }
    acc += __shfl_xor(acc, 1);
    acc += __shfl_xor(acc, 2);
    if ((tid & 3) == 0) yb[row*(H_*P_) + h*P_ + p] = acc;
    __syncthreads();
  }
}

// ---------------------------------------------------------------- fuse + bf16
__global__ __launch_bounds__(256)
void k_fuse(const float* __restrict__ proj, const float* __restrict__ yb,
            const float* __restrict__ D, __bf16* __restrict__ ybf)
{
  const int row = blockIdx.x;
  const int tid = threadIdx.x;
  const float* pr = proj + (long)row * DIN;
#pragma unroll
  for (int j = 0; j < 16; j++) {
    int idx = tid + j * 256;
    float x = pr[OFF_X + idx];
    float z = pr[OFF_Z + idx];
    float y = yb[(long)row*(H_*P_) + idx] + D[idx >> 6] * x;
    y *= z / (1.f + expf(-z));
    ybf[(long)row*(H_*P_) + idx] = (__bf16)y;
  }
}

// ---------------------------------------------------------------- launch
extern "C" void kernel_launch(void* const* d_in, const int* in_sizes, int n_in,
                              void* d_out, int out_size, void* d_ws, size_t ws_size,
                              hipStream_t stream)
{
  (void)in_sizes; (void)n_in; (void)out_size; (void)ws_size;
  const float* hs  = (const float*)d_in[0];
  const float* w1  = (const float*)d_in[1];
  const float* dtb = (const float*)d_in[2];
  const float* Bb  = (const float*)d_in[3];
  const float* Cb  = (const float*)d_in[4];
  const float* Bnw = (const float*)d_in[5];
  const float* Cnw = (const float*)d_in[6];
  const float* Dp  = (const float*)d_in[7];
  const float* w2  = (const float*)d_in[8];

  char* ws = (char*)d_ws;
  size_t off = 0;
  auto alloc = [&](size_t bytes) -> void* {
    off = (off + 255) & ~(size_t)255;
    void* p = ws + off; off += bytes; return p;
  };
  __bf16* hsbf = (__bf16*)alloc((size_t)ROWS * HID * 2);
  __bf16* w1bf = (__bf16*)alloc((size_t)DIN  * HID * 2);
  __bf16* w2bf = (__bf16*)alloc((size_t)HID  * (H_*P_) * 2);
  float*  proj = (float*)alloc((size_t)ROWS * DIN * 4);
  float*  DTb  = (float*)alloc((size_t)ROWS * H_ * 4);
  float*  decb = (float*)alloc((size_t)ROWS * H_ * 4);
  float*  lamb = (float*)alloc((size_t)ROWS * H_ * 4);
  float*  ub   = (float*)alloc((size_t)ROWS * H_ * P_ * 4);
  float*  Brms = (float*)alloc((size_t)ROWS * N_ * 4);
  float*  Crms = (float*)alloc((size_t)ROWS * N_ * 4);
  float*  yb   = (float*)alloc((size_t)ROWS * H_ * P_ * 4);
  __bf16* ybf  = (__bf16*)alloc((size_t)ROWS * H_ * P_ * 2);

  k_f32_to_bf16<<<1024, 256, 0, stream>>>(hs, hsbf, (long)ROWS * HID);
  k_f32_to_bf16<<<1024, 256, 0, stream>>>(w1, w1bf, (long)DIN  * HID);
  k_f32_to_bf16<<<1024, 256, 0, stream>>>(w2, w2bf, (long)HID * H_ * P_);

  dim3 g1((DIN + 127) / 128, ROWS / 128);
  k_gemm_bf16<<<g1, 256, 0, stream>>>(hsbf, w1bf, proj, ROWS, DIN, HID);

  k_prep<<<ROWS, 256, 0, stream>>>(proj, dtb, Bnw, Cnw, DTb, decb, lamb, ub, Brms, Crms);
  k_scan<<<B_ * H_, 256, 0, stream>>>(proj, DTb, decb, lamb, ub, Brms, Crms, Bb, Cb, yb);
  k_fuse<<<ROWS, 256, 0, stream>>>(proj, yb, Dp, ybf);

  dim3 g2((HID + 127) / 128, ROWS / 128);
  k_gemm_bf16<<<g2, 256, 0, stream>>>(ybf, w2bf, (float*)d_out, ROWS, HID, H_ * P_);
}
